// Base_48241072668906
// MI455X (gfx1250) — matrix operand order repaired and checked by reading, not compiled, not run on hardware
//
#include <hip/hip_runtime.h>
#include <hip/hip_bf16.h>

typedef __attribute__((ext_vector_type(16))) _Float16 v16h;
typedef __attribute__((ext_vector_type(8)))  _Float16 v8h;
typedef __attribute__((ext_vector_type(4)))  _Float16 v4h;
typedef __attribute__((ext_vector_type(8)))  float    v8f;

#define D_MODEL 1024
#define N_HEADS 16
#define DH      64
#define N_LAYERS 4
#define N_LEVELS 8
#define BB 2
#define TT 1024
#define ROWS (BB*TT)
#define EPSV 1e-5f
#define FMAXV 3.402823466e38f
#define S3D (3 * D_MODEL)

__device__ __forceinline__ v16h cat8(v8h lo, v8h hi) {
    return __builtin_shufflevector(lo, hi, 0,1,2,3,4,5,6,7,8,9,10,11,12,13,14,15);
}

__device__ __forceinline__ v8f wmma_f16(v16h a, v16h b, v8f c) {
    // v_wmma_f32_16x16x32_f16: (neg_a, A, neg_b, B, c_mod, C, reuse_a, reuse_b)
    return __builtin_amdgcn_wmma_f32_16x16x32_f16(false, a, false, b, (short)0, c, false, false);
}

// ---------------------------------------------------------------------------
// AdaLN: per-row layernorm stats + 2*(1-0.1*hn)*hn, exp(lg)*h + beta, * mask
// ---------------------------------------------------------------------------
__global__ void __launch_bounds__(256)
adaln_kernel(const float* __restrict__ X, const float* __restrict__ emb,
             const int* __restrict__ lvl, const float* __restrict__ mask,
             float* __restrict__ H)
{
    const int row = blockIdx.x;           // b*TT + t
    const int b   = row / TT;
    const int t   = threadIdx.x;
    const float* xr = X + (size_t)row * D_MODEL;

    float xv[4];
    float s = 0.f;
#pragma unroll
    for (int i = 0; i < 4; ++i) { xv[i] = xr[t + 256*i]; s += xv[i]; }

    __shared__ float red[256];
    red[t] = s; __syncthreads();
    for (int off = 128; off > 0; off >>= 1) {
        if (t < off) red[t] += red[t + off];
        __syncthreads();
    }
    const float mu = red[0] * (1.0f / D_MODEL);
    __syncthreads();

    float s2 = 0.f;
#pragma unroll
    for (int i = 0; i < 4; ++i) { float d = xv[i] - mu; s2 += d * d; }
    red[t] = s2; __syncthreads();
    for (int off = 128; off > 0; off >>= 1) {
        if (t < off) red[t] += red[t + off];
        __syncthreads();
    }
    const float var  = red[0] * (1.0f / D_MODEL);
    const float rinv = rsqrtf(var + EPSV);

    const float* g = emb + (size_t)lvl[b] * (2 * D_MODEL);
    const float mrow = mask[row];
#pragma unroll
    for (int i = 0; i < 4; ++i) {
        const int c = t + 256*i;
        const float hn = (xv[i] - mu) * rinv;
        const float h2 = 2.0f * (1.0f - 0.1f * hn) * hn;
        H[(size_t)row * D_MODEL + c] = (expf(g[c]) * h2 + g[D_MODEL + c]) * mrow;
    }
}

// ---------------------------------------------------------------------------
// WMMA GEMM: Y = epilogue(X[Mx KK] @ W[KKx NN]).  128x64 block tile, 4 waves
// of 64x32 (4x2 WMMA tiles = 8 wmma / K-step), K staged 32 at a time as f16
// in double-buffered LDS with software pipelining (1 barrier / step).
// NN/KK are compile-time so all staging addresses fold into load immediates.
// MODE 0: plain; 1: +bias, exact GELU; 2: (+bias+res)*mask.
// ---------------------------------------------------------------------------
template<int MODE, int NN, int KK>
__global__ void __launch_bounds__(128)
gemm_kernel(const float* __restrict__ X, const float* __restrict__ W,
            const float* __restrict__ bias, const float* __restrict__ res,
            const float* __restrict__ mask, float* __restrict__ Y)
{
    __shared__ _Float16 Ash[2][128 * 40];   // [row][k], pitch 40 halfs
    __shared__ _Float16 Bsh[2][64 * 40];    // [n][k] (transposed), pitch 40

    const int tid  = threadIdx.x;
    const int wave = tid >> 5;
    const int lane = tid & 31;
    const int ln   = lane & 15;
    const int lh   = lane >> 4;
    const int rowBlk = blockIdx.y * 128;
    const int colBlk = blockIdx.x * 64;
    const int wm = (wave >> 1) * 64;
    const int wn = (wave & 1) * 32;

    // staging thread mapping
    const int atk = (tid & 7) * 4;   // A k offset
    const int atr = tid >> 3;        // A row base (0..15), rows atr + 16p
    const int bn  = tid & 63;        // B n
    const int bk0 = tid >> 6;        // B k base (0..1), rows bk0 + 2p

    // per-thread staging base pointers, advanced by a constant per K-step so
    // the per-row offsets become compile-time load immediates
    const float* aPtr = X + (size_t)(rowBlk + atr) * KK + atk;
    const float* bPtr = W + (size_t)bk0 * NN + colBlk + bn;

    float4 aReg[8];
    float  bReg[16];

    auto loadTile = [&]() {
#pragma unroll
        for (int p = 0; p < 8; ++p)
            aReg[p] = *(const float4*)(aPtr + (size_t)p * 16 * KK);
#pragma unroll
        for (int p = 0; p < 16; ++p)
            bReg[p] = bPtr[(size_t)p * 2 * NN];
        aPtr += 32;
        bPtr += 32 * NN;
    };
    auto storeTile = [&](int buf) {
#pragma unroll
        for (int p = 0; p < 8; ++p) {
            v4h hv; hv.x = (_Float16)aReg[p].x; hv.y = (_Float16)aReg[p].y;
                    hv.z = (_Float16)aReg[p].z; hv.w = (_Float16)aReg[p].w;
            *(v4h*)(&Ash[buf][(atr + p * 16) * 40 + atk]) = hv;
        }
#pragma unroll
        for (int p = 0; p < 16; ++p)
            Bsh[buf][bn * 40 + bk0 + p * 2] = (_Float16)bReg[p];
    };

    v8f acc[4][2] = {};

    loadTile();
    storeTile(0);
    __syncthreads();

    const int nsteps = KK >> 5;
    for (int s = 0; s < nsteps; ++s) {
        const int p = s & 1;
        if (s + 1 < nsteps) loadTile();   // prefetch next tile to regs

        v16h af[4], bf[2];
#pragma unroll
        for (int mt = 0; mt < 4; ++mt) {
            const _Float16* base = &Ash[p][(wm + mt * 16 + ln) * 40];
            af[mt] = cat8(*(const v8h*)(base + 8 * lh),
                          *(const v8h*)(base + 16 + 8 * lh));
        }
#pragma unroll
        for (int nt = 0; nt < 2; ++nt) {
            const _Float16* base = &Bsh[p][(wn + nt * 16 + ln) * 40 + 8 * lh];
            bf[nt] = cat8(*(const v8h*)base, *(const v8h*)(base + 16));
        }
#pragma unroll
        for (int mt = 0; mt < 4; ++mt)
#pragma unroll
            for (int nt = 0; nt < 2; ++nt)
                acc[mt][nt] = wmma_f16(af[mt], bf[nt], acc[mt][nt]);

        if (s + 1 < nsteps) storeTile(p ^ 1);
        __syncthreads();
    }

    // epilogue
#pragma unroll
    for (int mt = 0; mt < 4; ++mt)
#pragma unroll
        for (int nt = 0; nt < 2; ++nt) {
            const int col = colBlk + wn + nt * 16 + ln;
            const float bv = (MODE == 0) ? 0.f : bias[col];
#pragma unroll
            for (int v = 0; v < 8; ++v) {
                const int row = rowBlk + wm + mt * 16 + v + 8 * lh;
                float val = acc[mt][nt][v] + bv;
                if (MODE == 1)
                    val = 0.5f * val * (1.0f + erff(val * 0.70710678118654752f));
                if (MODE == 2)
                    val = (res[(size_t)row * NN + col] + val) * mask[row];
                Y[(size_t)row * NN + col] = val;
            }
        }
}

// ---------------------------------------------------------------------------
// Flash-style attention.  Block = (q-supertile of 64, head, batch), 4 waves,
// each wave a 16-row query tile with dh=64 f32 accumulator (4 WMMA C tiles).
// Key loop in steps of 32 with K/V staged in LDS via constant-stride base
// pointers; P re-laid-out via LDS; next tile prefetched via global_prefetch.
// ---------------------------------------------------------------------------
__global__ void __launch_bounds__(128)
attn_kernel(const float* __restrict__ QKV, const float* __restrict__ mask,
            float* __restrict__ O)
{
    const int blk = blockIdx.x;     // 0..15 query supertile
    const int hd  = blockIdx.y;     // head
    const int b   = blockIdx.z;     // batch
    const int tid = threadIdx.x;
    const int wave = tid >> 5, lane = tid & 31, ln = lane & 15, lh = lane >> 4;
    const int i0 = blk * 64 + wave * 16;

    __shared__ _Float16 Ksh[32 * 72];      // [key][dh]   pitch 72
    __shared__ _Float16 Vsh[64 * 40];      // [dh][key]   pitch 40 (transposed)
    __shared__ _Float16 Psh[4 * 16 * 40];  // per-wave P  [row][key] pitch 40
    __shared__ float    kmsh[32];

    const size_t rowBase = (size_t)b * TT;
    const float* Qp = QKV + rowBase * S3D + (size_t)hd * DH;
    const float* Kp = Qp + D_MODEL;
    const float* Vp = Qp + 2 * D_MODEL;

    // Q fragments (A-matrix layout), dh split in two 32-chunks
    v16h qa[2];
#pragma unroll
    for (int f = 0; f < 2; ++f) {
        const float* qrow = Qp + (size_t)(i0 + ln) * S3D + 32 * f + 8 * lh;
        v16h tv;
#pragma unroll
        for (int e = 0; e < 8; ++e) tv[e] = (_Float16)qrow[e];
#pragma unroll
        for (int e = 0; e < 8; ++e) tv[8 + e] = (_Float16)qrow[16 + e];
        qa[f] = tv;
    }

    v8f oacc[4] = {};
    float mstat[8], lstat[8];
#pragma unroll
    for (int v = 0; v < 8; ++v) { mstat[v] = -FMAXV; lstat[v] = 0.f; }

    // constant-stride staging pointers (advanced by 32*S3D per key step)
    const int c4 = (tid & 15) * 4;
    const int kb = tid >> 4;
    const int vc = tid & 63;
    const int vk0 = tid >> 6;
    const float* kStage = Kp + (size_t)kb * S3D + c4;
    const float* vStage = Vp + (size_t)vk0 * S3D + vc;
    const float* kPref  = Kp + (size_t)((tid >> 2) + 32) * S3D + (tid & 3) * 16;
    const float* vPref  = Vp + (size_t)((tid >> 2) + 32) * S3D + (tid & 3) * 16;
    const float* mStage = mask + rowBase + tid;

    _Float16* myP = &Psh[wave * 16 * 40];
    const int smax = blk * 2 + 1;

    for (int s = 0; s <= smax; ++s) {
        const int j0 = s * 32;
        __syncthreads();
        // stage K (32 keys x 64 dh)
#pragma unroll
        for (int p = 0; p < 4; ++p) {
            const float4 f4 = *(const float4*)(kStage + (size_t)p * 8 * S3D);
            v4h hv; hv.x = (_Float16)f4.x; hv.y = (_Float16)f4.y;
                    hv.z = (_Float16)f4.z; hv.w = (_Float16)f4.w;
            *(v4h*)(&Ksh[(kb + p * 8) * 72 + c4]) = hv;
        }
        // stage V transposed (dh x keys)
#pragma unroll
        for (int p = 0; p < 16; ++p)
            Vsh[vc * 40 + vk0 + p * 2] = (_Float16)vStage[(size_t)p * 2 * S3D];
        if (tid < 32) kmsh[tid] = *mStage;
        // prefetch next key tile into cache while this one is consumed
        if (s < smax) {
            __builtin_prefetch(kPref, 0, 1);
            __builtin_prefetch(vPref, 0, 1);
        }
        kStage += 32 * S3D; vStage += 32 * S3D;
        kPref  += 32 * S3D; vPref  += 32 * S3D;
        mStage += 32;
        __syncthreads();

        if (i0 + 15 < j0) continue;   // fully above diagonal (wave-uniform)

        // S = Q K^T for two 16-key tiles
        v8f sacc[2] = {};
#pragma unroll
        for (int nt = 0; nt < 2; ++nt)
#pragma unroll
            for (int f = 0; f < 2; ++f) {
                const _Float16* base = &Ksh[(nt * 16 + ln) * 72 + 32 * f + 8 * lh];
                v16h kf = cat8(*(const v8h*)base, *(const v8h*)(base + 16));
                sacc[nt] = wmma_f16(qa[f], kf, sacc[nt]);
            }

        // mask, scale, online softmax stats
        float pvals[2][8], mnew[8];
#pragma unroll
        for (int v = 0; v < 8; ++v) {
            const int i = i0 + v + 8 * lh;
            float best = -FMAXV;
#pragma unroll
            for (int nt = 0; nt < 2; ++nt) {
                const int j = j0 + nt * 16 + ln;
                float sv = sacc[nt][v] * 0.125f;    // dh^-0.5
                const bool valid = (j <= i) && (kmsh[nt * 16 + ln] != 0.f);
                sv = valid ? sv : -FMAXV;
                pvals[nt][v] = sv;
                best = fmaxf(best, sv);
            }
#pragma unroll
            for (int off = 1; off < 16; off <<= 1)
                best = fmaxf(best, __shfl_xor(best, off, 16));
            mnew[v] = fmaxf(mstat[v], best);
        }

#pragma unroll
        for (int v = 0; v < 8; ++v) {
            float rs = 0.f;
            const int prow = v + 8 * lh;
#pragma unroll
            for (int nt = 0; nt < 2; ++nt) {
                const float p = expf(pvals[nt][v] - mnew[v]);
                rs += p;
                myP[prow * 40 + nt * 16 + ln] = (_Float16)p;
            }
#pragma unroll
            for (int off = 1; off < 16; off <<= 1)
                rs += __shfl_xor(rs, off, 16);
            const float alpha = expf(mstat[v] - mnew[v]);
            lstat[v] = lstat[v] * alpha + rs;
            mstat[v] = mnew[v];
#pragma unroll
            for (int dt = 0; dt < 4; ++dt) oacc[dt][v] *= alpha;
        }

        // re-read P as A fragment (wave-private LDS, DS ops in-order per wave)
        v16h pa;
        {
            const _Float16* base = &myP[ln * 40 + 8 * lh];
            pa = cat8(*(const v8h*)base, *(const v8h*)(base + 16));
        }
        // O += P @ V
#pragma unroll
        for (int dt = 0; dt < 4; ++dt) {
            const _Float16* base = &Vsh[(dt * 16 + ln) * 40 + 8 * lh];
            v16h vf = cat8(*(const v8h*)base, *(const v8h*)(base + 16));
            oacc[dt] = wmma_f16(pa, vf, oacc[dt]);
        }
    }

    float* Op = O + (rowBase + i0) * D_MODEL + (size_t)hd * DH;
#pragma unroll
    for (int v = 0; v < 8; ++v) {
        const float inv = 1.0f / lstat[v];
#pragma unroll
        for (int dt = 0; dt < 4; ++dt)
            Op[(size_t)(v + 8 * lh) * D_MODEL + dt * 16 + ln] = oacc[dt][v] * inv;
    }
}

// ---------------------------------------------------------------------------
extern "C" void kernel_launch(void* const* d_in, const int* in_sizes, int n_in,
                              void* d_out, int out_size, void* d_ws, size_t ws_size,
                              hipStream_t stream) {
    (void)in_sizes; (void)n_in; (void)out_size; (void)ws_size;
    const float* x        = (const float*)d_in[0];
    const float* m        = (const float*)d_in[1];
    const int*   lvl      = (const int*)  d_in[2];
    const float* Wqkv     = (const float*)d_in[3];
    const float* Wout     = (const float*)d_in[4];
    const float* bout     = (const float*)d_in[5];
    const float* ada_attn = (const float*)d_in[6];
    const float* ada_ffn  = (const float*)d_in[7];
    const float* W1       = (const float*)d_in[8];
    const float* b1       = (const float*)d_in[9];
    const float* W2       = (const float*)d_in[10];
    const float* b2       = (const float*)d_in[11];

    float* X = (float*)d_out;                         // resident activations
    float* h   = (float*)d_ws;                        // ROWS * D
    float* qkv = h   + (size_t)ROWS * D_MODEL;        // ROWS * 3D
    float* o   = qkv + (size_t)ROWS * 3 * D_MODEL;    // ROWS * D
    float* u   = o   + (size_t)ROWS * D_MODEL;        // ROWS * 4D

    hipMemcpyAsync(X, x, sizeof(float) * (size_t)ROWS * D_MODEL,
                   hipMemcpyDeviceToDevice, stream);

    const dim3 gQKV(3 * D_MODEL / 64, ROWS / 128);
    const dim3 gOUT(D_MODEL / 64, ROWS / 128);
    const dim3 gFF1(4 * D_MODEL / 64, ROWS / 128);
    const dim3 gFF2(D_MODEL / 64, ROWS / 128);

    for (int i = 0; i < N_LAYERS; ++i) {
        // attention sub-block
        adaln_kernel<<<ROWS, 256, 0, stream>>>(
            X, ada_attn + (size_t)i * N_LEVELS * 2 * D_MODEL, lvl, m, h);
        gemm_kernel<0, 3 * D_MODEL, D_MODEL><<<gQKV, 128, 0, stream>>>(
            h, Wqkv + (size_t)i * D_MODEL * 3 * D_MODEL,
            nullptr, nullptr, nullptr, qkv);
        attn_kernel<<<dim3(TT / 64, N_HEADS, BB), 128, 0, stream>>>(qkv, m, o);
        gemm_kernel<2, D_MODEL, D_MODEL><<<gOUT, 128, 0, stream>>>(
            o, Wout + (size_t)i * D_MODEL * D_MODEL,
            bout + (size_t)i * D_MODEL, X, m, X);
        // FFN sub-block
        adaln_kernel<<<ROWS, 256, 0, stream>>>(
            X, ada_ffn + (size_t)i * N_LEVELS * 2 * D_MODEL, lvl, m, h);
        gemm_kernel<1, 4 * D_MODEL, D_MODEL><<<gFF1, 128, 0, stream>>>(
            h, W1 + (size_t)i * D_MODEL * 4 * D_MODEL,
            b1 + (size_t)i * 4 * D_MODEL, nullptr, nullptr, u);
        gemm_kernel<2, D_MODEL, 4 * D_MODEL><<<gFF2, 128, 0, stream>>>(
            u, W2 + (size_t)i * 4 * D_MODEL * D_MODEL,
            b2 + (size_t)i * D_MODEL, X, m, X);
    }
}
